// TransformerModel_44598940402060
// MI455X (gfx1250) — compile-verified, matrix operand order checked by reading, not run on hardware
//
#include <hip/hip_runtime.h>
#include <hip/hip_bf16.h>

typedef __attribute__((ext_vector_type(16))) _Float16 v16h;
typedef __attribute__((ext_vector_type(8)))  _Float16 h8;
typedef __attribute__((ext_vector_type(8)))  float    v8f;

#define BB   128
#define LSEQ 250
#define DM   384
#define NH   8
#define DH   48
#define NLAY 4
#define HIDC 342
#define HPAD 344      // f16 row stride for the 342-wide tensor (multiple of 8)
#define C2LD 256      // padded row stride for c2 weight (250 -> 256)
#define UTOP 30

// ---------------------------------------------------------------------------
// WMMA GEMM v4 (f16 operands, async A staging, split K tail):
//   C[batch](f32) = (acc ? C : 0) + Ashift[batch](f16) @ Bshift[batch](f16)
//   lda/ldb: element row strides (lda multiple of 8 for 16B-aligned staging).
//   rowShift: A row l -> l+rowShift within rowPeriod (zero outside)
//   colShift: B column n -> n+colShift within [0,N) (zero outside)
// Block 256 threads = 8 waves; tile 64x64, K panel 64. Wave w: 16x32 slab
// (wm=(w&3)*16, wn=(w>>2)*32) -> 4 v_wmma per wave per staging round.
// A tile staged with GLOBAL_LOAD_ASYNC_TO_LDS_B128 (ASYNCcnt), B tile staged
// with lane-coalesced strided reads + contiguous 16B LDS stores (transpose).
// ---------------------------------------------------------------------------
__global__ void gemm_wmma(const _Float16* __restrict__ A, long aBS, int lda,
                          const _Float16* __restrict__ Bm, long bBS, int ldb,
                          float* __restrict__ C, long cBS,
                          int M, int N, int K,
                          int rowShift, int rowPeriod, int colShift,
                          int accumulate)
{
    __shared__ _Float16 sA[64][72];   // (m,k)
    __shared__ _Float16 sB[64][72];   // transposed (n,k)

    const _Float16* Ab = A  + (long)blockIdx.z * aBS;
    const _Float16* Bb = Bm + (long)blockIdx.z * bBS;
    float*          Cb = C  + (long)blockIdx.z * cBS;

    int tileM = blockIdx.x * 64;
    int tileN = blockIdx.y * 64;
    int tid   = threadIdx.x;
    int lane  = tid & 31;
    int wave  = tid >> 5;
    int wm    = (wave & 3) * 16;
    int wn    = (wave >> 2) * 32;

    // ---- A staging assignment: one row segment (16 halfs = 32 B) per thread
    int rA   = tid >> 2;              // 0..63
    int segA = (tid & 3) * 16;        // 0,16,32,48
    int gmA  = tileM + rA;
    int rowA = -1;
    if (gmA < M) {
        rowA = gmA;
        if (rowShift != 0) {
            int l = gmA % rowPeriod;
            int ls = l + rowShift;
            rowA = (ls < 0 || ls >= rowPeriod) ? -1 : (gmA - l + ls);
        }
    }
    unsigned ldsA = (unsigned)(uintptr_t)&sA[rA][segA];   // LDS byte offset

    // ---- B staging assignment: one column segment per thread ----
    int nB   = tid & 63;
    int segB = (tid >> 6) * 16;       // 0,16,32,48
    int gnB  = tileN + nB;
    int colB = gnB + colShift;
    bool bok = (gnB < N) && (colB >= 0) && (colB < N);

    v8f acc0 = {}, acc1 = {};

    int half  = lane >> 4;
    int mrow  = wm + (lane & 15);
    int ncol0 = wn + (lane & 15);

    auto compute_panel = [&]() {
#pragma unroll
        for (int ks = 0; ks < 64; ks += 32) {
            v16h af, bf0, bf1;
#pragma unroll
            for (int v = 0; v < 8; ++v) {
                int kb = ks + (v < 4 ? 2 * v : 16 + 2 * (v - 4)) + 8 * half;
                af[2 * v]      = sA[mrow][kb];
                af[2 * v + 1]  = sA[mrow][kb + 1];
                bf0[2 * v]     = sB[ncol0][kb];
                bf0[2 * v + 1] = sB[ncol0][kb + 1];
                bf1[2 * v]     = sB[ncol0 + 16][kb];
                bf1[2 * v + 1] = sB[ncol0 + 16][kb + 1];
            }
            acc0 = __builtin_amdgcn_wmma_f32_16x16x32_f16(
                       false, af, false, bf0, (short)0, acc0, false, false);
            acc1 = __builtin_amdgcn_wmma_f32_16x16x32_f16(
                       false, af, false, bf1, (short)0, acc1, false, false);
        }
    };

    const h8 hz = {};
    int kMain = K & ~63;              // full 64-panels

    for (int kk = 0; kk < kMain; kk += 64) {
        // ---- stage A: async global->LDS copy (32 B per lane) ----
        if (rowA >= 0) {
            const _Float16* ap = Ab + (long)rowA * lda + kk + segA;
            asm volatile("global_load_async_to_lds_b128 %0, %1, off"
                         :: "v"(ldsA), "v"(ap) : "memory");
            asm volatile("global_load_async_to_lds_b128 %0, %1, off offset:16"
                         :: "v"(ldsA), "v"(ap) : "memory");
        } else {
            *(h8*)&sA[rA][segA]     = hz;
            *(h8*)&sA[rA][segA + 8] = hz;
        }

        // ---- stage B transposed (coalesced strided reads, 16B LDS stores)
        if (bok) {
            const _Float16* bp = Bb + (long)(kk + segB) * ldb + colB;
            alignas(16) _Float16 tmp[16];
#pragma unroll
            for (int e = 0; e < 16; ++e) tmp[e] = bp[(long)e * ldb];
            *(h8*)&sB[nB][segB]     = *(const h8*)&tmp[0];
            *(h8*)&sB[nB][segB + 8] = *(const h8*)&tmp[8];
        } else {
            *(h8*)&sB[nB][segB]     = hz;
            *(h8*)&sB[nB][segB + 8] = hz;
        }

        if (kk + 64 < K)
            __builtin_prefetch(&Bb[(long)(kk + 64) * ldb + tileN], 0, 0);

        asm volatile("s_wait_asynccnt 0x0" ::: "memory");
        __syncthreads();
        compute_panel();
        __syncthreads();
    }

    // ---- K tail panel (guarded scalar staging) ----
    if (kMain < K) {
        int kk = kMain;
#pragma unroll
        for (int e = 0; e < 16; ++e) {
            int gk = kk + segA + e;
            _Float16 v = (_Float16)0.f;
            if (rowA >= 0 && gk < K) v = Ab[(long)rowA * lda + gk];
            sA[rA][segA + e] = v;
        }
#pragma unroll
        for (int e = 0; e < 16; ++e) {
            int gk = kk + segB + e;
            _Float16 v = (_Float16)0.f;
            if (bok && gk < K) v = Bb[(long)gk * ldb + colB];
            sB[nB][segB + e] = v;
        }
        __syncthreads();
        compute_panel();
        __syncthreads();
    }

    // D layout: VGPR r: lanes0-15 -> M=r, lanes16-31 -> M=r+8; N = lane&15
    int n0 = tileN + wn + (lane & 15);
    int n1 = n0 + 16;
#pragma unroll
    for (int r = 0; r < 8; ++r) {
        int m = tileM + wm + r + 8 * half;
        if (m < M) {
            if (n0 < N) {
                long idx = (long)m * N + n0;
                float v = acc0[r];
                if (accumulate) v += Cb[idx];
                Cb[idx] = v;
            }
            if (n1 < N) {
                long idx = (long)m * N + n1;
                float v = acc1[r];
                if (accumulate) v += Cb[idx];
                Cb[idx] = v;
            }
        }
    }
}

// ---------------------------------------------------------------------------
// Elementwise / reduction helpers
// ---------------------------------------------------------------------------
__device__ __forceinline__ float gelu_f(float x) {
    const float c = 0.7978845608028654f;
    float t = tanhf(c * (x + 0.044715f * x * x * x));
    return 0.5f * x * (1.f + t);
}

__global__ void f32_to_f16(const float* __restrict__ src, _Float16* __restrict__ dst, long n)
{
    long i = (long)blockIdx.x * blockDim.x + threadIdx.x;
    if (i >= n) return;
    dst[i] = (_Float16)src[i];
}

// max-pool 1000->250, LN, +pos.  grid = B*LSEQ, block = 128 (f32 out)
__global__ void pool_ln_pos(const float* __restrict__ x, const float* __restrict__ pos,
                            const float* __restrict__ w, const float* __restrict__ b,
                            float* __restrict__ out)
{
    int row = blockIdx.x;
    int bb = row / LSEQ, s = row % LSEQ;
    int t = threadIdx.x;
    const float* base = x + ((long)bb * 1000 + s * 4) * DM;
    float v[3];
#pragma unroll
    for (int i = 0; i < 3; ++i) {
        int d = t + i * 128;
        float m = base[d];
        m = fmaxf(m, base[DM + d]);
        m = fmaxf(m, base[2 * DM + d]);
        m = fmaxf(m, base[3 * DM + d]);
        v[i] = m;
    }
    __shared__ float red[128];
    red[t] = v[0] + v[1] + v[2]; __syncthreads();
    for (int o = 64; o > 0; o >>= 1) { if (t < o) red[t] += red[t + o]; __syncthreads(); }
    float mu = red[0] / (float)DM; __syncthreads();
    float d0 = v[0] - mu, d1 = v[1] - mu, d2 = v[2] - mu;
    red[t] = d0 * d0 + d1 * d1 + d2 * d2; __syncthreads();
    for (int o = 64; o > 0; o >>= 1) { if (t < o) red[t] += red[t + o]; __syncthreads(); }
    float inv = rsqrtf(red[0] / (float)DM + 1e-5f);
    long ob = (long)row * DM;
    out[ob + t]       = d0 * inv * w[t]       + b[t]       + pos[(long)s * DM + t];
    out[ob + t + 128] = d1 * inv * w[t + 128] + b[t + 128] + pos[(long)s * DM + t + 128];
    out[ob + t + 256] = d2 * inv * w[t + 256] + b[t + 256] + pos[(long)s * DM + t + 256];
}

// LayerNorm over 384 cols, fp32 out. grid = rows, block = 128
__global__ void layernorm384(const float* __restrict__ in, const float* __restrict__ w,
                             const float* __restrict__ b, float* __restrict__ out)
{
    long row = blockIdx.x;
    const float* p = in + row * DM;
    int t = threadIdx.x;
    float v0 = p[t], v1 = p[t + 128], v2 = p[t + 256];
    __shared__ float red[128];
    red[t] = v0 + v1 + v2; __syncthreads();
    for (int o = 64; o > 0; o >>= 1) { if (t < o) red[t] += red[t + o]; __syncthreads(); }
    float mu = red[0] / (float)DM; __syncthreads();
    float d0 = v0 - mu, d1 = v1 - mu, d2 = v2 - mu;
    red[t] = d0 * d0 + d1 * d1 + d2 * d2; __syncthreads();
    for (int o = 64; o > 0; o >>= 1) { if (t < o) red[t] += red[t + o]; __syncthreads(); }
    float inv = rsqrtf(red[0] / (float)DM + 1e-5f);
    out[row * DM + t]       = d0 * inv * w[t]       + b[t];
    out[row * DM + t + 128] = d1 * inv * w[t + 128] + b[t + 128];
    out[row * DM + t + 256] = d2 * inv * w[t + 256] + b[t + 256];
}

// LayerNorm over 384 cols, f16 out (feeds WMMA GEMMs). grid = rows, block = 128
__global__ void layernorm384_f16(const float* __restrict__ in, const float* __restrict__ w,
                                 const float* __restrict__ b, _Float16* __restrict__ out)
{
    long row = blockIdx.x;
    const float* p = in + row * DM;
    int t = threadIdx.x;
    float v0 = p[t], v1 = p[t + 128], v2 = p[t + 256];
    __shared__ float red[128];
    red[t] = v0 + v1 + v2; __syncthreads();
    for (int o = 64; o > 0; o >>= 1) { if (t < o) red[t] += red[t + o]; __syncthreads(); }
    float mu = red[0] / (float)DM; __syncthreads();
    float d0 = v0 - mu, d1 = v1 - mu, d2 = v2 - mu;
    red[t] = d0 * d0 + d1 * d1 + d2 * d2; __syncthreads();
    for (int o = 64; o > 0; o >>= 1) { if (t < o) red[t] += red[t + o]; __syncthreads(); }
    float inv = rsqrtf(red[0] / (float)DM + 1e-5f);
    out[row * DM + t]       = (_Float16)(d0 * inv * w[t]       + b[t]);
    out[row * DM + t + 128] = (_Float16)(d1 * inv * w[t + 128] + b[t + 128]);
    out[row * DM + t + 256] = (_Float16)(d2 * inv * w[t + 256] + b[t + 256]);
}

__global__ void bias_act(float* __restrict__ y, const float* __restrict__ bias,
                         int N, long total, int do_gelu)
{
    long i = (long)blockIdx.x * blockDim.x + threadIdx.x;
    if (i >= total) return;
    float v = y[i] + bias[i % N];
    if (do_gelu) v = gelu_f(v);
    y[i] = v;
}

// f16 gelu epilogue with padded output stride (342 -> 344)
__global__ void bias_gelu_f16_pad(const float* __restrict__ src, const float* __restrict__ bias,
                                  _Float16* __restrict__ dst, long rows)
{
    long total = rows * HPAD;
    long i = (long)blockIdx.x * blockDim.x + threadIdx.x;
    if (i >= total) return;
    long m = i / HPAD;
    int c = (int)(i % HPAD);
    float v = 0.f;
    if (c < HIDC) v = gelu_f(src[m * HIDC + c] + bias[c]);
    dst[i] = (_Float16)v;
}

__global__ void bias_add_residual(float* __restrict__ x, const float* __restrict__ y,
                                  const float* __restrict__ bias, int N, long total)
{
    long i = (long)blockIdx.x * blockDim.x + threadIdx.x;
    if (i >= total) return;
    x[i] += y[i] + bias[i % N];
}

// x += gelu(y + bias[seq-channel])
__global__ void c2_epilogue(float* __restrict__ x, const float* __restrict__ y,
                            const float* __restrict__ bias, long total)
{
    long i = (long)blockIdx.x * blockDim.x + threadIdx.x;
    if (i >= total) return;
    int s = (int)((i / DM) % LSEQ);
    x[i] += gelu_f(y[i] + bias[s]);
}

// pack grouped conv weight (O, Ig, 3) -> 3 dense f16 K x O slices (as GEMM B)
__global__ void pack_conv_f16(const float* __restrict__ w, _Float16* __restrict__ dst,
                              int O, int Ig, int groups, int K)
{
    long tot = 3L * K * O;
    long i = (long)blockIdx.x * blockDim.x + threadIdx.x;
    if (i >= tot) return;
    int n = (int)(i % O);
    long tk = i / O;
    int k = (int)(tk % K);
    int t = (int)(tk / K);
    int og = O / groups;
    int g = n / og;
    int ii = k - g * Ig;
    float v = 0.f;
    if (ii >= 0 && ii < Ig) v = w[((long)n * Ig + ii) * 3 + t];
    dst[i] = (_Float16)v;
}

// pack grouped conv weight as GEMM-A (row-major M x K, padded stride ldk)
__global__ void pack_conv_rowA_f16(const float* __restrict__ w, _Float16* __restrict__ dst,
                                   int O, int Ig, int groups, int K, int ldk)
{
    long tot = 3L * O * ldk;
    long i = (long)blockIdx.x * blockDim.x + threadIdx.x;
    if (i >= tot) return;
    int k = (int)(i % ldk);
    long tn = i / ldk;
    int n = (int)(tn % O);         // out channel = A row
    int t = (int)(tn / O);
    int og = O / groups;
    int g = n / og;
    int ii = k - g * Ig;
    float v = 0.f;
    if (k < K && ii >= 0 && ii < Ig) v = w[((long)n * Ig + ii) * 3 + t];
    dst[i] = (_Float16)v;
    (void)t;
}

// ---------------------------------------------------------------------------
// ProbSparse attention pieces (scalar; ~2% of FLOPs)
// ---------------------------------------------------------------------------
__global__ void vmean_kernel(const float* __restrict__ v, float* __restrict__ vmean)
{
    int b = blockIdx.x;
    int hd = threadIdx.x;        // 384
    float s = 0.f;
    for (int l = 0; l < LSEQ; ++l) s += v[((long)(b * LSEQ + l)) * DM + hd];
    vmean[b * DM + hd] = s * (1.f / (float)LSEQ);
}

__global__ void msc_kernel(const float* __restrict__ q, const float* __restrict__ k,
                           float* __restrict__ Msc, int layer)
{
    int bh = blockIdx.x;         // B*NH
    int l = threadIdx.x;
    if (l >= LSEQ) return;
    int b = bh >> 3, h = bh & 7;
    const float* qp = q + ((long)(b * LSEQ + l)) * DM + h * DH;
    float mx = -1e30f, sm = 0.f;
    for (int u = 0; u < UTOP; ++u) {
        unsigned hsh = 1103515245u * ((unsigned)l * 31u + (unsigned)u * 2654435761u
                                      + (unsigned)layer * 97u) + 12345u;
        int li = (int)(hsh % (unsigned)LSEQ);
        const float* kp = k + ((long)(b * LSEQ + li)) * DM + h * DH;
        float d = 0.f;
        for (int c = 0; c < DH; ++c) d += qp[c] * kp[c];
        mx = fmaxf(mx, d);
        sm += d;
    }
    Msc[(long)bh * LSEQ + l] = mx - sm * (1.f / (float)UTOP);
}

__global__ void topk_kernel(const float* __restrict__ Msc, int* __restrict__ top)
{
    int bh = blockIdx.x;
    int t = threadIdx.x;
    __shared__ float vals[LSEQ];
    __shared__ float sv[256];
    __shared__ int   si[256];
    if (t < LSEQ) vals[t] = Msc[(long)bh * LSEQ + t];
    __syncthreads();
    for (int u = 0; u < UTOP; ++u) {
        sv[t] = (t < LSEQ) ? vals[t] : -1e38f;
        si[t] = t;
        __syncthreads();
        for (int o = 128; o > 0; o >>= 1) {
            if (t < o && sv[t + o] > sv[t]) { sv[t] = sv[t + o]; si[t] = si[t + o]; }
            __syncthreads();
        }
        if (t == 0) { top[bh * UTOP + u] = si[0]; vals[si[0]] = -1e38f; }
        __syncthreads();
    }
}

__global__ void scores_softmax(const float* __restrict__ q, const float* __restrict__ k,
                               const int* __restrict__ top, float* __restrict__ attn)
{
    int idx = blockIdx.x;                 // b*240 + h*30 + u
    int u = idx % UTOP;
    int h = (idx / UTOP) & 7;
    int b = idx / (UTOP * NH);
    int l = threadIdx.x;
    __shared__ float red[256];
    int tq = top[(b * NH + h) * UTOP + u];
    const float* qp = q + ((long)(b * LSEQ + tq)) * DM + h * DH;
    float s = -1e30f;
    if (l < LSEQ) {
        const float* kp = k + ((long)(b * LSEQ + l)) * DM + h * DH;
        float d = 0.f;
        for (int c = 0; c < DH; ++c) d += qp[c] * kp[c];
        s = d * 0.14433756729740643f;     // 1/sqrt(48)
    }
    red[l] = s; __syncthreads();
    for (int o = 128; o > 0; o >>= 1) { if (l < o) red[l] = fmaxf(red[l], red[l + o]); __syncthreads(); }
    float mx = red[0]; __syncthreads();
    float e = (l < LSEQ) ? __expf(s - mx) : 0.f;
    red[l] = e; __syncthreads();
    for (int o = 128; o > 0; o >>= 1) { if (l < o) red[l] += red[l + o]; __syncthreads(); }
    float inv = 1.f / red[0];
    if (l < LSEQ) attn[(long)idx * LSEQ + l] = e * inv;
}

// context buffer is f16 (it feeds the O-projection GEMM)
__global__ void ctx_fill_f16(const float* __restrict__ vmean, _Float16* __restrict__ ctx)
{
    long total = (long)BB * LSEQ * DM;
    long i = (long)blockIdx.x * blockDim.x + threadIdx.x;
    if (i >= total) return;
    int d = (int)(i % DM);
    int b = (int)(i / ((long)LSEQ * DM));
    ctx[i] = (_Float16)vmean[b * DM + d];
}

__global__ void ctx_scatter_f16(const float* __restrict__ attn, const float* __restrict__ v,
                                const int* __restrict__ top, _Float16* __restrict__ ctx)
{
    int idx = blockIdx.x;
    int dh = threadIdx.x;
    if (dh >= DH) return;
    int u = idx % UTOP;
    int h = (idx / UTOP) & 7;
    int b = idx / (UTOP * NH);
    const float* ap = attn + (long)idx * LSEQ;
    float s = 0.f;
    for (int l = 0; l < LSEQ; ++l)
        s += ap[l] * v[((long)(b * LSEQ + l)) * DM + h * DH + dh];
    int tl = top[(b * NH + h) * UTOP + u];
    ctx[((long)(b * LSEQ + tl)) * DM + h * DH + dh] = (_Float16)s;
}

// ---------------------------------------------------------------------------
// Final head
// ---------------------------------------------------------------------------
__global__ void maxpool_L(const float* __restrict__ in, float* __restrict__ out)
{
    int b = blockIdx.x;
    int d = threadIdx.x;     // 384
    float m = -1e30f;
    for (int l = 0; l < LSEQ; ++l)
        m = fmaxf(m, in[((long)(b * LSEQ + l)) * DM + d]);
    out[b * DM + d] = m;
}

__global__ void fc_kernel(const float* __restrict__ in, const float* __restrict__ w,
                          const float* __restrict__ bias, float* __restrict__ out)
{
    int b = blockIdx.x;
    int c = threadIdx.x;
    if (c >= 10) return;
    float s = bias[c];
    for (int d = 0; d < DM; ++d) s += in[b * DM + d] * w[d * 10 + c];
    out[b * 10 + c] = s;
}

// ---------------------------------------------------------------------------
// Host orchestration
// ---------------------------------------------------------------------------
static void launch_gemm(hipStream_t st,
                        const _Float16* A, long aBS, int lda,
                        const _Float16* Bm, long bBS, int ldb,
                        float* C, long cBS, int M, int N, int K, int batch,
                        int rowShift, int rowPeriod, int colShift, int acc)
{
    dim3 g((M + 63) / 64, (N + 63) / 64, batch);
    gemm_wmma<<<g, 256, 0, st>>>(A, aBS, lda, Bm, bBS, ldb, C, cBS, M, N, K,
                                 rowShift, rowPeriod, colShift, acc);
}

extern "C" void kernel_launch(void* const* d_in, const int* in_sizes, int n_in,
                              void* d_out, int out_size, void* d_ws, size_t ws_size,
                              hipStream_t stream)
{
    (void)in_sizes; (void)n_in; (void)out_size; (void)ws_size;
    const float* x       = (const float*)d_in[0];
    const float* pos     = (const float*)d_in[1];
    const float* ln_in_w = (const float*)d_in[2];
    const float* ln_in_b = (const float*)d_in[3];
    const float* lnA_w   = (const float*)d_in[4];
    const float* lnA_b   = (const float*)d_in[5];
    const float* Wq      = (const float*)d_in[6];
    const float* bq      = (const float*)d_in[7];
    const float* Wk      = (const float*)d_in[8];
    const float* bk      = (const float*)d_in[9];
    const float* Wv      = (const float*)d_in[10];
    const float* bv      = (const float*)d_in[11];
    const float* Wo      = (const float*)d_in[12];
    const float* bo      = (const float*)d_in[13];
    const float* lnB_w   = (const float*)d_in[14];
    const float* lnB_b   = (const float*)d_in[15];
    const float* c1a_w   = (const float*)d_in[16];
    const float* c1a_b   = (const float*)d_in[17];
    const float* c1b_w   = (const float*)d_in[18];
    const float* c1b_b   = (const float*)d_in[19];
    const float* lnC_w   = (const float*)d_in[20];
    const float* lnC_b   = (const float*)d_in[21];
    const float* c2_w    = (const float*)d_in[22];
    const float* c2_b    = (const float*)d_in[23];
    const float* ln2_w   = (const float*)d_in[24];
    const float* ln2_b   = (const float*)d_in[25];
    const float* fc_w    = (const float*)d_in[26];
    const float* fc_b    = (const float*)d_in[27];
    float* out = (float*)d_out;

    const long XE   = (long)BB * LSEQ * DM;        // 12,288,000
    const long ROWS = (long)BB * LSEQ;             // 32000
    const long TOTAL = ROWS * DM;
    const long WSZ  = (long)NLAY * DM * DM;        // per projection weight array

    char* ws = (char*)d_ws;
    size_t off = 0;
    auto allocB = [&](size_t bytes) -> void* {
        void* p = (void*)(ws + off);
        off += (bytes + 255) & ~(size_t)255;
        return p;
    };
    // fp32 buffers
    float* xbuf   = (float*)allocB(XE * 4);
    float* qbuf   = (float*)allocB(XE * 4);
    float* kbuf   = (float*)allocB(XE * 4);
    float* vbuf   = (float*)allocB(XE * 4);
    float* cbufF  = (float*)allocB(XE * 4);
    float* tbufF  = (float*)allocB(XE * 4);
    float* attnb  = (float*)allocB((size_t)BB * NH * UTOP * LSEQ * 4);
    float* Mscb   = (float*)allocB((size_t)BB * NH * LSEQ * 4);
    float* vmeanb = (float*)allocB((size_t)BB * DM * 4);
    float* pooled = (float*)allocB((size_t)BB * DM * 4);
    float* pooled2= (float*)allocB((size_t)BB * DM * 4);
    int*   topb   = (int*)allocB((size_t)BB * NH * UTOP * 4);
    // f16 buffers
    _Float16* hbuf16  = (_Float16*)allocB(XE * 2);
    _Float16* cbuf16  = (_Float16*)allocB(XE * 2);
    _Float16* tbuf16  = (_Float16*)allocB((size_t)ROWS * HPAD * 2);
    _Float16* wq16    = (_Float16*)allocB((size_t)WSZ * 2);
    _Float16* wk16    = (_Float16*)allocB((size_t)WSZ * 2);
    _Float16* wv16    = (_Float16*)allocB((size_t)WSZ * 2);
    _Float16* wo16    = (_Float16*)allocB((size_t)WSZ * 2);
    _Float16* packw16 = (_Float16*)allocB((size_t)3 * DM * DM * 2);
    _Float16* packc2  = (_Float16*)allocB((size_t)3 * LSEQ * C2LD * 2);

    const int EB = 256;
    auto blocks = [&](long n) { return (unsigned)((n + EB - 1) / EB); };

    // convert projection weights to f16 once (all layers)
    f32_to_f16<<<blocks(WSZ), EB, 0, stream>>>(Wq, wq16, WSZ);
    f32_to_f16<<<blocks(WSZ), EB, 0, stream>>>(Wk, wk16, WSZ);
    f32_to_f16<<<blocks(WSZ), EB, 0, stream>>>(Wv, wv16, WSZ);
    f32_to_f16<<<blocks(WSZ), EB, 0, stream>>>(Wo, wo16, WSZ);

    // input: max-pool + LN + pos
    pool_ln_pos<<<(unsigned)ROWS, 128, 0, stream>>>(x, pos, ln_in_w, ln_in_b, xbuf);

    for (int l = 0; l < NLAY; ++l) {
        const _Float16* wq = wq16 + (long)l * DM * DM;
        const _Float16* wk = wk16 + (long)l * DM * DM;
        const _Float16* wv = wv16 + (long)l * DM * DM;
        const _Float16* wo = wo16 + (long)l * DM * DM;

        // ---- ProbSparse attention ----
        layernorm384_f16<<<(unsigned)ROWS, 128, 0, stream>>>(xbuf, lnA_w + l * DM, lnA_b + l * DM, hbuf16);
        launch_gemm(stream, hbuf16, 0, DM, wq, 0, DM, qbuf, 0, (int)ROWS, DM, DM, 1, 0, 1, 0, 0);
        bias_act<<<blocks(TOTAL), EB, 0, stream>>>(qbuf, bq + l * DM, DM, TOTAL, 0);
        launch_gemm(stream, hbuf16, 0, DM, wk, 0, DM, kbuf, 0, (int)ROWS, DM, DM, 1, 0, 1, 0, 0);
        bias_act<<<blocks(TOTAL), EB, 0, stream>>>(kbuf, bk + l * DM, DM, TOTAL, 0);
        launch_gemm(stream, hbuf16, 0, DM, wv, 0, DM, vbuf, 0, (int)ROWS, DM, DM, 1, 0, 1, 0, 0);
        bias_act<<<blocks(TOTAL), EB, 0, stream>>>(vbuf, bv + l * DM, DM, TOTAL, 0);

        vmean_kernel<<<BB, DM, 0, stream>>>(vbuf, vmeanb);
        msc_kernel<<<BB * NH, 256, 0, stream>>>(qbuf, kbuf, Mscb, l);
        topk_kernel<<<BB * NH, 256, 0, stream>>>(Mscb, topb);
        scores_softmax<<<BB * NH * UTOP, 256, 0, stream>>>(qbuf, kbuf, topb, attnb);
        ctx_fill_f16<<<blocks(TOTAL), EB, 0, stream>>>(vmeanb, cbuf16);
        ctx_scatter_f16<<<BB * NH * UTOP, 64, 0, stream>>>(attnb, vbuf, topb, cbuf16);

        launch_gemm(stream, cbuf16, 0, DM, wo, 0, DM, tbufF, 0, (int)ROWS, DM, DM, 1, 0, 1, 0, 0);
        bias_add_residual<<<blocks(TOTAL), EB, 0, stream>>>(xbuf, tbufF, bo + l * DM, DM, TOTAL);

        // ---- grouped conv FFN: c1a (gelu) + c1b, 3 shifted GEMMs each ----
        layernorm384_f16<<<(unsigned)ROWS, 128, 0, stream>>>(xbuf, lnB_w + l * DM, lnB_b + l * DM, hbuf16);
        pack_conv_f16<<<blocks(3L * DM * HIDC), EB, 0, stream>>>(c1a_w + (long)l * HIDC * 128 * 3,
                                                                 packw16, HIDC, 128, 3, DM);
        for (int t = 0; t < 3; ++t)
            launch_gemm(stream, hbuf16, 0, DM, packw16 + (long)t * DM * HIDC, 0, HIDC,
                        tbufF, 0, (int)ROWS, HIDC, DM, 1, t - 1, LSEQ, 0, t > 0);
        bias_gelu_f16_pad<<<blocks(ROWS * HPAD), EB, 0, stream>>>(tbufF, c1a_b + l * HIDC, tbuf16, ROWS);

        pack_conv_f16<<<blocks(3L * HIDC * DM), EB, 0, stream>>>(c1b_w + (long)l * DM * 114 * 3,
                                                                 packw16, DM, 114, 3, HIDC);
        for (int t = 0; t < 3; ++t)
            launch_gemm(stream, tbuf16, 0, HPAD, packw16 + (long)t * HIDC * DM, 0, DM,
                        cbufF, 0, (int)ROWS, DM, HIDC, 1, t - 1, LSEQ, 0, t > 0);
        bias_add_residual<<<blocks(TOTAL), EB, 0, stream>>>(xbuf, cbufF, c1b_b + l * DM, DM, TOTAL);

        // ---- c2: sequence-channel conv over d-axis (per-batch GEMM) ----
        layernorm384_f16<<<(unsigned)ROWS, 128, 0, stream>>>(xbuf, lnC_w + l * DM, lnC_b + l * DM, hbuf16);
        pack_conv_rowA_f16<<<blocks(3L * LSEQ * C2LD), EB, 0, stream>>>(c2_w + (long)l * LSEQ * 125 * 3,
                                                                        packc2, LSEQ, 125, 2, LSEQ, C2LD);
        for (int t = 0; t < 3; ++t)
            launch_gemm(stream, packc2 + (long)t * LSEQ * C2LD, 0, C2LD,
                        hbuf16, (long)LSEQ * DM, DM,
                        cbufF, (long)LSEQ * DM,
                        LSEQ, DM, LSEQ, BB, 0, 1, t - 1, t > 0);
        c2_epilogue<<<blocks(TOTAL), EB, 0, stream>>>(xbuf, cbufF, c2_b + l * LSEQ, TOTAL);
    }

    // ---- head: LN -> max over L -> LN -> FC ----
    layernorm384<<<(unsigned)ROWS, 128, 0, stream>>>(xbuf, ln2_w, ln2_b, tbufF);
    maxpool_L<<<BB, DM, 0, stream>>>(tbufF, pooled);
    layernorm384<<<BB, 128, 0, stream>>>(pooled, ln2_w, ln2_b, pooled2);
    fc_kernel<<<BB, 32, 0, stream>>>(pooled2, fc_w, fc_b, out);
}
